// KAIST_GAT_89318139887644
// MI455X (gfx1250) — matrix operand order repaired and checked by reading, not compiled, not run on hardware
//
#include <hip/hip_runtime.h>

// ---------------- problem constants (from reference) ----------------
#define NN    30000
#define EE    240000
#define ETOT  270000   // E + N self loops
#define GG    128
#define DH    128
#define HEADS 4

typedef __bf16 bf16;
typedef __attribute__((ext_vector_type(16))) __bf16 v16bf;
typedef __attribute__((ext_vector_type(8)))  float  v8f;

// A-matrix (16x32 bf16) per-lane K index for fragment element i (contiguous: k = 16*hi + i, matches B)
__device__ __forceinline__ int a_kidx(int i, bool hi) {
    return (hi ? 16 : 0) + i;
}

__device__ __forceinline__ void atomic_max_f32(float* addr, float val) {
    int* ai = (int*)addr;
    int cur = __float_as_int(*addr);
    while (__int_as_float(cur) < val) {
        int old = atomicCAS(ai, cur, __float_as_int(val));
        if (old == cur) break;
        cur = old;
    }
}

// -------- per-layer zero/init: hmid accumulator, seg-max, seg-sum, BN sums ----
__global__ void k_init_layer(float* hmid, float* m, float* z, float* ssum, float* ssq) {
    int i = blockIdx.x * blockDim.x + threadIdx.x;
    if (i < NN * DH)    hmid[i] = 0.f;
    if (i < NN * HEADS) { m[i] = -3.0e38f; z[i] = 0.f; }
    if (i < DH)         { ssum[i] = 0.f; ssq[i] = 0.f; }
}

// -------- GEMM: Hb[N,512](bf16) = hin[N,128](f32->bf16) @ W[128,512] ----------
// grid (N/16, 4), block 256 (8 waves); wave w -> 16-col tile within 128-col block
// B staged TRANSPOSED in LDS ([col][k]) so B-fragment loads are contiguous b128s.
__global__ __launch_bounds__(256) void k_gemm_hw(const float* __restrict__ hin,
                                                 const float* __restrict__ W,
                                                 bf16* __restrict__ Hb) {
    __shared__ bf16 As[16][128];
    __shared__ bf16 Bsw[128][128];   // [out-col j][k]
    int tid  = threadIdx.x;
    int row0 = blockIdx.x * 16;
    int col0 = blockIdx.y * 128;
    for (int i = tid; i < 16 * 128; i += 256) {
        int r = i >> 7, c = i & 127;
        As[r][c] = (bf16)hin[(size_t)(row0 + r) * DH + c];
    }
    for (int i = tid; i < 128 * 128; i += 256) {
        int k = i >> 7, j = i & 127;          // j fast-varying -> coalesced W read
        Bsw[j][k] = (bf16)W[(size_t)k * 512 + col0 + j];
    }
    __syncthreads();
    int lane = tid & 31, wave = tid >> 5;
    int n0   = wave * 16;
    bool hi  = lane >= 16;
    int mrow = lane & 15, ncol = lane & 15;
    v8f acc = {};
#pragma unroll
    for (int k0 = 0; k0 < 128; k0 += 32) {
        v16bf a, b;
#pragma unroll
        for (int i = 0; i < 16; ++i) a[i] = As[mrow][k0 + a_kidx(i, hi)];
        int kb = k0 + (hi ? 16 : 0);
#pragma unroll
        for (int i = 0; i < 16; ++i) b[i] = Bsw[n0 + ncol][kb + i];
        acc = __builtin_amdgcn_wmma_f32_16x16x32_bf16(false, a, false, b, (short)0, acc, false, false);
    }
#pragma unroll
    for (int i = 0; i < 8; ++i) {
        int r = i + (hi ? 8 : 0);
        Hb[(size_t)(row0 + r) * 512 + col0 + n0 + ncol] = (bf16)acc[i];
    }
}

// -------- attention logits: al_s/al_d [N,H] = <Hb[n,h,:], att[h,:]> ----------
__global__ void k_att_logits(const bf16* __restrict__ Hb,
                             const float* __restrict__ asrc, const float* __restrict__ adst,
                             float* __restrict__ als, float* __restrict__ ald) {
    int i = blockIdx.x * blockDim.x + threadIdx.x;
    if (i >= NN * HEADS) return;
    int n = i >> 2, h = i & 3;
    const bf16* hp = Hb + (size_t)n * 512 + h * 128;
    const float* as = asrc + h * 128;
    const float* ad = adst + h * 128;
    float s0 = 0.f, s1 = 0.f;
#pragma unroll 8
    for (int c = 0; c < 128; ++c) {
        float v = (float)hp[c];
        s0 += v * as[c]; s1 += v * ad[c];
    }
    als[i] = s0; ald[i] = s1;
}

// -------- edge pass 1: leaky_relu logit + segment max over dst ---------------
__global__ void k_edge_max(const int* __restrict__ ei, const float* __restrict__ als,
                           const float* __restrict__ ald, float* __restrict__ eatt,
                           float* __restrict__ m) {
    int e = blockIdx.x * blockDim.x + threadIdx.x;
    if (e >= ETOT) return;
    int src = (e < EE) ? ei[e]      : (e - EE);
    int dst = (e < EE) ? ei[EE + e] : (e - EE);
#pragma unroll
    for (int h = 0; h < HEADS; ++h) {
        float v = als[src * 4 + h] + ald[dst * 4 + h];
        v = (v > 0.f) ? v : 0.2f * v;
        eatt[(size_t)e * 4 + h] = v;
        atomic_max_f32(&m[dst * 4 + h], v);
    }
}

// -------- edge pass 2: exp(e - max) + segment sum ----------------------------
__global__ void k_edge_exp(const int* __restrict__ ei, const float* __restrict__ m,
                           float* __restrict__ eatt, float* __restrict__ z) {
    int e = blockIdx.x * blockDim.x + threadIdx.x;
    if (e >= ETOT) return;
    int dst = (e < EE) ? ei[EE + e] : (e - EE);
#pragma unroll
    for (int h = 0; h < HEADS; ++h) {
        float t = expf(eatt[(size_t)e * 4 + h] - m[dst * 4 + h]);
        eatt[(size_t)e * 4 + h] = t;
        atomicAdd(&z[dst * 4 + h], t);
    }
}

// -------- edge pass 3: hmid[dst,c] += 0.25 * sum_h alpha_h * Hb[src,h,c] -----
__global__ __launch_bounds__(128) void k_edge_msg(const int* __restrict__ ei,
                                                  const float* __restrict__ exv,
                                                  const float* __restrict__ z,
                                                  const bf16* __restrict__ Hb,
                                                  float* __restrict__ hmid) {
    int e = blockIdx.x;
    int c = threadIdx.x;
    int src = (e < EE) ? ei[e]      : (e - EE);
    int dst = (e < EE) ? ei[EE + e] : (e - EE);
    __shared__ float sal[HEADS];
    if (c < HEADS) sal[c] = exv[(size_t)e * 4 + c] / (z[dst * 4 + c] + 1e-16f);
    const bf16* hp = Hb + (size_t)src * 512;
    __builtin_prefetch(hp + c, 0, 1);   // global_prefetch_b8: pull source row
    __syncthreads();
    float v = 0.25f * (sal[0] * (float)hp[c]       + sal[1] * (float)hp[128 + c] +
                       sal[2] * (float)hp[256 + c] + sal[3] * (float)hp[384 + c]);
    atomicAdd(&hmid[(size_t)dst * 128 + c], v);
}

// -------- BN stats pass: add conv bias in place, accumulate sum/sumsq --------
__global__ __launch_bounds__(128) void k_bn_stats(float* __restrict__ hmid,
                                                  const float* __restrict__ conv_b,
                                                  float* __restrict__ ssum, float* __restrict__ ssq) {
    int c  = threadIdx.x;
    int r0 = blockIdx.x * 100;
    float b = conv_b[c];
    float s = 0.f, q = 0.f;
    for (int r = r0; r < r0 + 100; ++r) {
        float v = hmid[(size_t)r * 128 + c] + b;
        hmid[(size_t)r * 128 + c] = v;
        s += v; q += v * v;
    }
    atomicAdd(&ssum[c], s);
    atomicAdd(&ssq[c], q);
}

__global__ void k_bn_final(const float* ssum, const float* ssq,
                           const float* gamma, const float* beta,
                           float* muv, float* gsv, float* bbv) {
    int c = threadIdx.x;
    float mu  = ssum[c] * (1.f / NN);
    float var = ssq[c] * (1.f / NN) - mu * mu;
    muv[c] = mu;
    gsv[c] = gamma[c] * rsqrtf(var + 1e-5f);
    bbv[c] = beta[c];
}

__global__ void k_bn_apply(float* hmid, const float* muv, const float* gsv, const float* bbv) {
    int i = blockIdx.x * blockDim.x + threadIdx.x;
    if (i >= NN * DH) return;
    int c = i & 127;
    hmid[i] = (hmid[i] - muv[c]) * gsv[c] + bbv[c];
}

// -------- gated skip: glin = hin@Win + y@Wout (+biases) via WMMA, fused epilogue
// grid N/16, block 256; hout may alias hin (safe: per-element read-then-write).
// B chunks staged TRANSPOSED ([col][k]) for contiguous fragment loads.
__global__ __launch_bounds__(256) void k_gate(const float* hin, const float* y,
                                              const float* __restrict__ Win,
                                              const float* __restrict__ Wout,
                                              const float* __restrict__ bin,
                                              const float* __restrict__ bout,
                                              float* hout) {
    __shared__ bf16 As1[16][128], As2[16][128];
    __shared__ bf16 Bs1[128][64], Bs2[128][64];   // [out-col j][k within 64-chunk]
    int tid  = threadIdx.x;
    int row0 = blockIdx.x * 16;
    for (int i = tid; i < 16 * 128; i += 256) {
        int r = i >> 7, c = i & 127;
        As1[r][c] = (bf16)hin[(size_t)(row0 + r) * 128 + c];
        As2[r][c] = (bf16)y  [(size_t)(row0 + r) * 128 + c];
    }
    int lane = tid & 31, wave = tid >> 5;
    int n0   = wave * 16;
    bool hi  = lane >= 16;
    int mrow = lane & 15, ncol = lane & 15;
    v8f acc = {};
    for (int kc = 0; kc < 2; ++kc) {
        __syncthreads();
        for (int i = tid; i < 64 * 128; i += 256) {
            int k = i >> 7, j = i & 127;      // j fast-varying -> coalesced reads
            Bs1[j][k] = (bf16)Win [(size_t)(kc * 64 + k) * 128 + j];
            Bs2[j][k] = (bf16)Wout[(size_t)(kc * 64 + k) * 128 + j];
        }
        __syncthreads();
#pragma unroll
        for (int k0 = 0; k0 < 64; k0 += 32) {
            v16bf a1, a2, b1, b2;
#pragma unroll
            for (int i = 0; i < 16; ++i) {
                int kk = a_kidx(i, hi);
                a1[i] = As1[mrow][kc * 64 + k0 + kk];
                a2[i] = As2[mrow][kc * 64 + k0 + kk];
            }
            int kb = k0 + (hi ? 16 : 0);
#pragma unroll
            for (int i = 0; i < 16; ++i) {
                b1[i] = Bs1[n0 + ncol][kb + i];
                b2[i] = Bs2[n0 + ncol][kb + i];
            }
            acc = __builtin_amdgcn_wmma_f32_16x16x32_bf16(false, a1, false, b1, (short)0, acc, false, false);
            acc = __builtin_amdgcn_wmma_f32_16x16x32_bf16(false, a2, false, b2, (short)0, acc, false, false);
        }
    }
#pragma unroll
    for (int i = 0; i < 8; ++i) {
        int r = row0 + i + (hi ? 8 : 0);
        int c = n0 + ncol;
        float glin = acc[i] + bin[c] + bout[c];
        float gate = 1.f / (1.f + expf(-glin));
        size_t idx = (size_t)r * 128 + c;
        float yv = y[idx], hv = hin[idx];
        hout[idx] = fmaxf(gate * yv + (1.f - gate) * hv, 0.f);
    }
}

// -------- global mean pool ---------------------------------------------------
__global__ void k_pool_zero(float* hg, float* cnt) {
    int i = blockIdx.x * blockDim.x + threadIdx.x;
    if (i < GG * DH) hg[i] = 0.f;
    if (i < GG)      cnt[i] = 0.f;
}
__global__ __launch_bounds__(128) void k_pool_sum(const float* __restrict__ h,
                                                  const int* __restrict__ batch,
                                                  float* __restrict__ hg, float* __restrict__ cnt) {
    int n = blockIdx.x, c = threadIdx.x;
    int g = batch[n];
    atomicAdd(&hg[g * 128 + c], h[(size_t)n * 128 + c]);
    if (c == 0) atomicAdd(&cnt[g], 1.0f);
}
__global__ void k_pool_div(float* hg, const float* cnt) {
    int i = blockIdx.x * blockDim.x + threadIdx.x;
    if (i >= GG * DH) return;
    hg[i] /= fmaxf(cnt[i >> 7], 1.0f);
}

// -------- head: relu(hg@fc1+b1) @ fc3 + b3, single block, WMMA ---------------
__global__ __launch_bounds__(256) void k_head(const float* __restrict__ hg,
                                              const float* __restrict__ fc1W, const float* __restrict__ fc1b,
                                              const float* __restrict__ fc3W, const float* __restrict__ fc3b,
                                              float* __restrict__ out) {
    __shared__ bf16 mid[128][128];
    int tid  = threadIdx.x;
    int lane = tid & 31, wave = tid >> 5;
    bool hi  = lane >= 16;
    int mrow = lane & 15, ncol = lane & 15;
    int r0   = wave * 16;
    for (int ct = 0; ct < 8; ++ct) {
        int n0 = ct * 16;
        v8f acc = {};
#pragma unroll
        for (int k0 = 0; k0 < 128; k0 += 32) {
            v16bf a, b;
#pragma unroll
            for (int i = 0; i < 16; ++i) a[i] = (bf16)hg[(size_t)(r0 + mrow) * 128 + k0 + a_kidx(i, hi)];
#pragma unroll
            for (int i = 0; i < 16; ++i) b[i] = (bf16)fc1W[(size_t)(k0 + (hi ? 16 : 0) + i) * 128 + n0 + ncol];
            acc = __builtin_amdgcn_wmma_f32_16x16x32_bf16(false, a, false, b, (short)0, acc, false, false);
        }
#pragma unroll
        for (int i = 0; i < 8; ++i) {
            int r = i + (hi ? 8 : 0);
            mid[r0 + r][n0 + ncol] = (bf16)fmaxf(acc[i] + fc1b[n0 + ncol], 0.f);
        }
    }
    __syncthreads();
    v8f acc = {};
#pragma unroll
    for (int k0 = 0; k0 < 128; k0 += 32) {
        v16bf a, b;
#pragma unroll
        for (int i = 0; i < 16; ++i) a[i] = mid[r0 + mrow][k0 + a_kidx(i, hi)];
#pragma unroll
        for (int i = 0; i < 16; ++i) b[i] = (bf16)fc3W[(size_t)(k0 + (hi ? 16 : 0) + i) * 16 + ncol];
        acc = __builtin_amdgcn_wmma_f32_16x16x32_bf16(false, a, false, b, (short)0, acc, false, false);
    }
#pragma unroll
    for (int i = 0; i < 8; ++i)
        out[(size_t)(r0 + i + (hi ? 8 : 0)) * 16 + ncol] = acc[i] + fc3b[ncol];
}

// ============================ host launcher ==================================
extern "C" void kernel_launch(void* const* d_in, const int* in_sizes, int n_in,
                              void* d_out, int out_size, void* d_ws, size_t ws_size,
                              hipStream_t stream) {
    const float* x      = (const float*)d_in[0];
    const int*   ei     = (const int*)  d_in[1];
    const int*   batch  = (const int*)  d_in[2];
    const float* W      = (const float*)d_in[3];   // [3,128,512]
    const float* asrc   = (const float*)d_in[4];   // [3,4,128]
    const float* adst   = (const float*)d_in[5];
    const float* conv_b = (const float*)d_in[6];   // [3,128]
    const float* gamma  = (const float*)d_in[7];
    const float* beta   = (const float*)d_in[8];
    const float* skWin  = (const float*)d_in[9];   // [3,128,128]
    const float* skbin  = (const float*)d_in[10];
    const float* skWout = (const float*)d_in[11];
    const float* skbout = (const float*)d_in[12];
    const float* fc1W   = (const float*)d_in[13];
    const float* fc1b   = (const float*)d_in[14];
    const float* fc3W   = (const float*)d_in[15];
    const float* fc3b   = (const float*)d_in[16];

    char* ws = (char*)d_ws;
    size_t off = 0;
    auto alloc = [&](size_t bytes) -> char* {
        char* p = ws + off;
        off = (off + bytes + 255) & ~(size_t)255;
        return p;
    };
    float* hcur = (float*)alloc((size_t)NN * DH * 4);
    bf16*  Hb   = (bf16*) alloc((size_t)NN * 512 * 2);
    float* hmid = (float*)alloc((size_t)NN * DH * 4);   // GAT out, then BN'd y
    float* als  = (float*)alloc((size_t)NN * HEADS * 4);
    float* ald  = (float*)alloc((size_t)NN * HEADS * 4);
    float* mm   = (float*)alloc((size_t)NN * HEADS * 4);
    float* zz   = (float*)alloc((size_t)NN * HEADS * 4);
    float* exv  = (float*)alloc((size_t)ETOT * HEADS * 4);
    float* ssum = (float*)alloc(DH * 4);
    float* ssq  = (float*)alloc(DH * 4);
    float* muv  = (float*)alloc(DH * 4);
    float* gsv  = (float*)alloc(DH * 4);
    float* bbv  = (float*)alloc(DH * 4);
    float* hg   = (float*)alloc((size_t)GG * DH * 4);
    float* cnt  = (float*)alloc(GG * 4);

    for (int l = 0; l < 3; ++l) {
        const float* hin = (l == 0) ? x : hcur;
        k_init_layer<<<(NN * DH + 255) / 256, 256, 0, stream>>>(hmid, mm, zz, ssum, ssq);
        dim3 g2(NN / 16, 4);
        k_gemm_hw<<<g2, 256, 0, stream>>>(hin, W + (size_t)l * DH * 512, Hb);
        k_att_logits<<<(NN * HEADS + 255) / 256, 256, 0, stream>>>(
            Hb, asrc + (size_t)l * HEADS * DH, adst + (size_t)l * HEADS * DH, als, ald);
        k_edge_max<<<(ETOT + 255) / 256, 256, 0, stream>>>(ei, als, ald, exv, mm);
        k_edge_exp<<<(ETOT + 255) / 256, 256, 0, stream>>>(ei, mm, exv, zz);
        k_edge_msg<<<ETOT, 128, 0, stream>>>(ei, exv, zz, Hb, hmid);
        k_bn_stats<<<NN / 100, 128, 0, stream>>>(hmid, conv_b + l * DH, ssum, ssq);
        k_bn_final<<<1, 128, 0, stream>>>(ssum, ssq, gamma + l * DH, beta + l * DH, muv, gsv, bbv);
        k_bn_apply<<<(NN * DH + 255) / 256, 256, 0, stream>>>(hmid, muv, gsv, bbv);
        k_gate<<<NN / 16, 256, 0, stream>>>(hin, hmid,
                                            skWin + (size_t)l * DH * DH, skWout + (size_t)l * DH * DH,
                                            skbin + l * DH, skbout + l * DH, hcur);
    }
    k_pool_zero<<<(GG * DH + 255) / 256, 256, 0, stream>>>(hg, cnt);
    k_pool_sum<<<NN, 128, 0, stream>>>(hcur, batch, hg, cnt);
    k_pool_div<<<(GG * DH + 255) / 256, 256, 0, stream>>>(hg, cnt);
    k_head<<<1, 256, 0, stream>>>(hg, fc1W, fc1b, fc3W, fc3b, (float*)d_out);
}
